// MyLayerFunc_53008486367609
// MI455X (gfx1250) — matrix operand order repaired and checked by reading, not compiled, not run on hardware
//
#include <hip/hip_runtime.h>

// ---------- types ----------
typedef __bf16 bhalf_t;
typedef __attribute__((ext_vector_type(16))) __bf16 v16bf;
typedef __attribute__((ext_vector_type(8)))  __bf16 v8bf;
typedef __attribute__((ext_vector_type(4)))  __bf16 v4bf;
typedef __attribute__((ext_vector_type(8)))  float  v8f;

// ---------- problem dims (match reference) ----------
#define BATCH 131072
#define KDIM  256
#define ODIM  1024

// ---------- tiling ----------
#define BM 128      // batch rows per block
#define BN 128      // out cols per block
#define BK 32       // K chunk (one bf16 WMMA K)
#define LDA 40      // padded LDS row stride (bf16 units) for A
#define LDB 40      // padded LDS row stride (bf16 units) for B

__device__ __forceinline__ v16bf cat8(v8bf lo, v8bf hi) {
  return __builtin_shufflevector(lo, hi, 0,1,2,3,4,5,6,7,8,9,10,11,12,13,14,15);
}

// ---------------------------------------------------------------------------
// Pass 1: split W (fp32) into hi/lo bf16 planes: w = hi + lo + O(2^-16)
// ---------------------------------------------------------------------------
__global__ __launch_bounds__(256)
void wsplit_kernel(const float* __restrict__ W,
                   unsigned short* __restrict__ whi,
                   unsigned short* __restrict__ wlo) {
  int i = blockIdx.x * 256 + threadIdx.x;          // grid covers ODIM*KDIM
  float w = W[i];
  bhalf_t h = (bhalf_t)w;
  bhalf_t l = (bhalf_t)(w - (float)h);
  whi[i] = __builtin_bit_cast(unsigned short, h);
  wlo[i] = __builtin_bit_cast(unsigned short, l);
}

// ---------------------------------------------------------------------------
// Pass 2: fused GEMM (bf16x3 split, fp32 accumulate) + per-column activation
// ---------------------------------------------------------------------------
__global__ __launch_bounds__(256)
void fused_gemm_act_kernel(const float* __restrict__ x,
                           const unsigned short* __restrict__ whi_u,
                           const unsigned short* __restrict__ wlo_u,
                           float* __restrict__ out) {
  const bhalf_t* __restrict__ whi = (const bhalf_t*)whi_u;
  const bhalf_t* __restrict__ wlo = (const bhalf_t*)wlo_u;

  __shared__ bhalf_t sAhi[BM * LDA];
  __shared__ bhalf_t sAlo[BM * LDA];
  __shared__ bhalf_t sBhi[BN * LDB];
  __shared__ bhalf_t sBlo[BN * LDB];

  const int tid   = threadIdx.x;
  const int lane  = tid & 31;
  const int wave  = tid >> 5;
  const int wm    = wave & 3;     // wave M position: rows wm*32 .. +31
  const int wn    = wave >> 2;    // wave N position: cols wn*64 .. +63
  const int lrow  = lane & 15;
  const int lhalf = lane >> 4;

  const int bm = (int)(blockIdx.x >> 3) * BM;   // 1024 row-bands
  const int bn = (int)(blockIdx.x & 7)  * BN;   // 8 col-bands (adjacent -> L2 reuse of x)

  v8f acc[2][4];
  #pragma unroll
  for (int mi = 0; mi < 2; ++mi)
    #pragma unroll
    for (int ni = 0; ni < 4; ++ni) { v8f z = {}; acc[mi][ni] = z; }

  // staging registers (software pipeline: global loads of chunk k+1 overlap WMMA of chunk k)
  float4 ra[4];
  v8bf   rbh[2], rbl[2];

  auto load_global = [&](int kk) {
    #pragma unroll
    for (int i = 0; i < 4; ++i) {            // A: 128 rows x 8 float4 per chunk
      int idx = tid + i * 256;
      int row = idx >> 3, c4 = idx & 7;
      ra[i] = *(const float4*)(x + (size_t)(bm + row) * KDIM + kk + c4 * 4);
    }
    #pragma unroll
    for (int i = 0; i < 2; ++i) {            // B: 128 rows x 4 v8bf per chunk (hi & lo)
      int idx = tid + i * 256;
      int row = idx >> 2, c = idx & 3;
      size_t off = (size_t)(bn + row) * KDIM + kk + c * 8;
      rbh[i] = *(const v8bf*)(whi + off);
      rbl[i] = *(const v8bf*)(wlo + off);
    }
  };

  auto store_lds = [&]() {
    #pragma unroll
    for (int i = 0; i < 4; ++i) {
      int idx = tid + i * 256;
      int row = idx >> 3, c4 = idx & 7;
      float f0 = ra[i].x, f1 = ra[i].y, f2 = ra[i].z, f3 = ra[i].w;
      bhalf_t h0 = (bhalf_t)f0, h1 = (bhalf_t)f1, h2 = (bhalf_t)f2, h3 = (bhalf_t)f3;
      v4bf h, l;
      h[0] = h0; h[1] = h1; h[2] = h2; h[3] = h3;
      l[0] = (bhalf_t)(f0 - (float)h0);
      l[1] = (bhalf_t)(f1 - (float)h1);
      l[2] = (bhalf_t)(f2 - (float)h2);
      l[3] = (bhalf_t)(f3 - (float)h3);
      *(v4bf*)(&sAhi[row * LDA + c4 * 4]) = h;   // ds_store_b64
      *(v4bf*)(&sAlo[row * LDA + c4 * 4]) = l;
    }
    #pragma unroll
    for (int i = 0; i < 2; ++i) {
      int idx = tid + i * 256;
      int row = idx >> 2, c = idx & 3;
      *(v8bf*)(&sBhi[row * LDB + c * 8]) = rbh[i];  // ds_store_b128
      *(v8bf*)(&sBlo[row * LDB + c * 8]) = rbl[i];
    }
  };

  load_global(0);

  for (int kc = 0; kc < KDIM / BK; ++kc) {
    store_lds();
    __syncthreads();
    if (kc < KDIM / BK - 1) load_global((kc + 1) * BK);

    // A fragments: lane holds row (lane&15); K halves {0..7,16..23} / {8..15,24..31} by lane>=16
    v16bf ah[2], al[2];
    #pragma unroll
    for (int mi = 0; mi < 2; ++mi) {
      int rb = (wm * 32 + mi * 16 + lrow) * LDA + lhalf * 8;
      ah[mi] = cat8(*(const v8bf*)&sAhi[rb], *(const v8bf*)&sAhi[rb + 16]);
      al[mi] = cat8(*(const v8bf*)&sAlo[rb], *(const v8bf*)&sAlo[rb + 16]);
    }
    // B fragments: lane holds col (lane&15) = W row o; K halves {0..7,16..23} / {8..15,24..31} by lane>=16
    #pragma unroll
    for (int ni = 0; ni < 4; ++ni) {
      int rb = (wn * 64 + ni * 16 + lrow) * LDB + lhalf * 8;
      v16bf bh = cat8(*(const v8bf*)&sBhi[rb], *(const v8bf*)&sBhi[rb + 16]);
      v16bf bl = cat8(*(const v8bf*)&sBlo[rb], *(const v8bf*)&sBlo[rb + 16]);
      #pragma unroll
      for (int mi = 0; mi < 2; ++mi) {
        acc[mi][ni] = __builtin_amdgcn_wmma_f32_16x16x32_bf16(
            false, ah[mi], false, bh, (short)0, acc[mi][ni], false, false);
        acc[mi][ni] = __builtin_amdgcn_wmma_f32_16x16x32_bf16(
            false, al[mi], false, bh, (short)0, acc[mi][ni], false, false);
        acc[mi][ni] = __builtin_amdgcn_wmma_f32_16x16x32_bf16(
            false, ah[mi], false, bl, (short)0, acc[mi][ni], false, false);
      }
    }
    __syncthreads();
  }

  // Epilogue: col%4 == lane&3 -> branchless activation select, coalesced 64B stores.
  // Matmul outputs are ~N(0, 0.58): hardware v_sin/v_cos (via __sinf/__cosf) are
  // accurate in this range and ~10x cheaper than the libm range-reduction path.
  const int act = lane & 3;
  #pragma unroll
  for (int mi = 0; mi < 2; ++mi) {
    #pragma unroll
    for (int ni = 0; ni < 4; ++ni) {
      const int    col     = bn + wn * 64 + ni * 16 + lrow;
      const size_t rowbase = (size_t)(bm + wm * 32 + mi * 16 + lhalf * 8);
      #pragma unroll
      for (int v = 0; v < 8; ++v) {
        float val = acc[mi][ni][v];
        float s = __sinf(val);                    // v_sin_f32
        float c = __cosf(val);                    // v_cos_f32
        float t = __fdividef(s, c);               // v_rcp_f32 + mul
        float r = (act == 0) ? val : (act == 1) ? c : (act == 2) ? s : t;
        out[(rowbase + v) * ODIM + col] = r;
      }
    }
  }
}

// ---------------------------------------------------------------------------
extern "C" void kernel_launch(void* const* d_in, const int* in_sizes, int n_in,
                              void* d_out, int out_size, void* d_ws, size_t ws_size,
                              hipStream_t stream) {
  const float* x = (const float*)d_in[0];     // [131072, 256]
  const float* W = (const float*)d_in[1];     // [1024, 256]
  float* out = (float*)d_out;                 // [131072, 1024]

  unsigned short* whi = (unsigned short*)d_ws;           // 512 KB
  unsigned short* wlo = whi + (size_t)ODIM * KDIM;       // 512 KB

  wsplit_kernel<<<(ODIM * KDIM) / 256, 256, 0, stream>>>(W, whi, wlo);

  dim3 grid((BATCH / BM) * (ODIM / BN));   // 1024 * 8 = 8192 blocks
  fused_gemm_act_kernel<<<grid, 256, 0, stream>>>(x, whi, wlo, out);
}
